// TalkingHeadAttention_36850819400466
// MI455X (gfx1250) — matrix operand order repaired and checked by reading, not compiled, not run on hardware
//
#include <hip/hip_runtime.h>
#include <hip/hip_bf16.h>

// ---------------- problem constants ----------------
constexpr int Bb   = 4;
constexpr int Ntok = 1024;
constexpr int Cc   = 1024;
constexpr int Hh   = 16;
constexpr int HDd  = 64;
constexpr int C3   = 3 * Cc;      // 3072
constexpr int BN   = Bb * Ntok;   // 4096

// ---------------- WMMA vector types ----------------
typedef __attribute__((ext_vector_type(16))) __bf16 v16bf;
typedef __attribute__((ext_vector_type(8)))  float  v8f;
typedef unsigned short u16;

__device__ __forceinline__ u16 bf16_bits(float x) {
    union { float f; unsigned int u; } c; c.f = x;
    // round-to-nearest-even fp32 -> bf16
    return (u16)((c.u + 0x7FFFu + ((c.u >> 16) & 1u)) >> 16);
}

// A fragment 16x32 (MxK) from bf16 row-major: per-lane data is
// one 32-byte run: K in [kb, kb+16)  (kb = (lane>>4)*16).
// p must already point at &src[m_lane*ld + kb].
__device__ __forceinline__ v16bf frag_ld_a(const u16* p) {
    union { v16bf v; uint4 q[2]; } u;
    u.q[0] = *(const uint4*)(p);
    u.q[1] = *(const uint4*)(p + 8);
    return u.v;
}

// B fragment 32x16 (KxN) from bf16 with K contiguous per n: per-lane data is
// one 32-byte run: K in [kb2, kb2+16)  (kb2 = (lane>>4)*16).
// p must already point at &src[n_lane*ld + kb2].
__device__ __forceinline__ v16bf frag_ld_b(const u16* p) {
    union { v16bf v; uint4 q[2]; } u;
    u.q[0] = *(const uint4*)(p);
    u.q[1] = *(const uint4*)(p + 8);
    return u.v;
}

#define WMMA_BF16(a, b, c) \
    __builtin_amdgcn_wmma_f32_16x16x32_bf16(false, (a), false, (b), (short)0, (c), false, false)

// ============================================================
// fp32 -> bf16 bulk convert (n must be a multiple of 1024; 4 elems/thread)
// ============================================================
__global__ void __launch_bounds__(256)
f32_to_bf16_kernel(const float* __restrict__ src, u16* __restrict__ dst) {
    const size_t i = ((size_t)blockIdx.x * 256 + threadIdx.x) * 4;
    float4 v = *(const float4*)(src + i);
    uint2 p;
    p.x = (unsigned)bf16_bits(v.x) | ((unsigned)bf16_bits(v.y) << 16);
    p.y = (unsigned)bf16_bits(v.z) | ((unsigned)bf16_bits(v.w) << 16);
    *(uint2*)(dst + i) = p;
}

// ============================================================
// GEMM: Y[M,Nn] = Xbf[M,K] @ Wbf^T + bias.  K compile-time.
// 256 thr = 8 waves -> 64(M) x 32(N) block tile (4x2 waves of 16x16).
// OUT_BF16 ? store bf16 : store fp32.
// ============================================================
template <int K, bool OUT_BF16>
__global__ void __launch_bounds__(256)
gemm_bf16(const u16* __restrict__ X, const u16* __restrict__ W,
          const float* __restrict__ bias, void* __restrict__ Yout, int Nn) {
    const int lane = threadIdx.x & 31;
    const int wave = threadIdx.x >> 5;
    const int m0 = blockIdx.x * 64 + (wave & 3) * 16;
    const int n0 = blockIdx.y * 32 + (wave >> 2) * 16;

    const u16* ap = X + (size_t)(m0 + (lane & 15)) * K + ((lane >> 4) * 16);
    const u16* bp = W + (size_t)(n0 + (lane & 15)) * K + ((lane >> 4) * 16);

    v8f acc = {};
#pragma unroll 4
    for (int kk = 0; kk < K; kk += 32) {
        __builtin_prefetch(ap + 128, 0, 3);
        __builtin_prefetch(bp + 128, 0, 3);
        acc = WMMA_BF16(frag_ld_a(ap), frag_ld_b(bp), acc);
        ap += 32; bp += 32;
    }
    const int nc = n0 + (lane & 15);
    const int mh = (lane >> 4) * 8;
    const float bv = bias[nc];
    if (OUT_BF16) {
        u16* Y = (u16*)Yout;
#pragma unroll
        for (int r = 0; r < 8; ++r)
            Y[(size_t)(m0 + mh + r) * Nn + nc] = bf16_bits(acc[r] + bv);
    } else {
        float* Y = (float*)Yout;
#pragma unroll
        for (int r = 0; r < 8; ++r)
            Y[(size_t)(m0 + mh + r) * Nn + nc] = acc[r] + bv;
    }
}

// ============================================================
// QK^T: attn[b,h,n,m] = (q.k) * scale, K-dim = HD = 64. grid (16,32,B*H)
// ============================================================
__global__ void __launch_bounds__(256)
qk_kernel(const u16* __restrict__ qkvbf, float* __restrict__ attn) {
    const int lane = threadIdx.x & 31;
    const int wave = threadIdx.x >> 5;
    const int bh = blockIdx.z;
    const int b = bh >> 4, h = bh & 15;

    const u16* base = qkvbf + (size_t)b * Ntok * C3;
    const int n0 = blockIdx.x * 64 + (wave & 3) * 16;   // query rows (M)
    const int m0 = blockIdx.y * 32 + (wave >> 2) * 16;  // key cols   (N)

    const u16* ap = base + h * HDd + (size_t)(n0 + (lane & 15)) * C3 + ((lane >> 4) * 16);
    const u16* bp = base + Cc + h * HDd + (size_t)(m0 + (lane & 15)) * C3 + ((lane >> 4) * 16);

    v8f acc = {};
#pragma unroll
    for (int kk = 0; kk < HDd; kk += 32)
        acc = WMMA_BF16(frag_ld_a(ap + kk), frag_ld_b(bp + kk), acc);

    float* out = attn + (size_t)bh * Ntok * Ntok;
    const int nc = m0 + (lane & 15);
    const int mh = (lane >> 4) * 8;
    const float scale = 0.125f;   // HD^-0.5 (applied post-accumulate: same linear op)
#pragma unroll
    for (int r = 0; r < 8; ++r)
        out[(size_t)(n0 + mh + r) * Ntok + nc] = acc[r] * scale;
}

// ============================================================
// Fused talking-heads: wl-mix -> softmax(m) -> ww-mix.
// One block per (b,n). Slab load uses CDNA5 async global->LDS (ASYNCcnt),
// bypassing the VGPR round-trip. Writes fp32 attn in place (d_out region)
// AND a bf16 copy for the A.V GEMM.
// ============================================================
__global__ void __launch_bounds__(256)
mix_softmax_mix(float* __restrict__ attn, u16* __restrict__ attnbf,
                const float* __restrict__ wl, const float* __restrict__ bl,
                const float* __restrict__ ww, const float* __restrict__ bw) {
    extern __shared__ float smem[];
    float* tile = smem;                 // 16*1024
    float* red  = tile + Hh * Ntok;     // 16*16
    float* wls  = red + 256;            // 256
    float* wws  = wls + 256;            // 256
    float* bls  = wws + 256;            // 16
    float* bws  = bls + 16;             // 16

    const int t  = threadIdx.x;
    const int bn = blockIdx.x;
    const int b  = bn >> 10;
    const int n  = bn & (Ntok - 1);
    const size_t plane = (size_t)Ntok * Ntok;
    float* basep  = attn   + (size_t)b * Hh * plane + (size_t)n * Ntok;
    u16*   basebf = attnbf + (size_t)b * Hh * plane + (size_t)n * Ntok;

    // --- async slab copy: thread t owns columns 4t..4t+3 of every head row.
    // 16 x global_load_async_to_lds_b128 per thread; LDS addr = low 32 bits of
    // the generic shared pointer (wave-relative LDS offset per ISA 10.2).
    {
        const unsigned lds0 = (unsigned)(uintptr_t)&tile[4 * t];
        const float* gp0 = basep + 4 * t;
#pragma unroll
        for (int r = 0; r < Hh; ++r) {
            unsigned lr = lds0 + (unsigned)(r * Ntok * 4);
            unsigned long long ga = (unsigned long long)(uintptr_t)(gp0 + (size_t)r * plane);
            asm volatile("global_load_async_to_lds_b128 %0, %1, off"
                         :: "v"(lr), "v"(ga) : "memory");
        }
    }

    wls[t] = wl[t];
    wws[t] = ww[t];
    if (t < Hh) { bls[t] = bl[t]; bws[t] = bw[t]; }

    asm volatile("s_wait_asynccnt 0x0" ::: "memory");
    __syncthreads();

    // mix 1 (wl,bl): column-owned, in place
#pragma unroll
    for (int c = 0; c < 4; ++c) {
        const int m = t + 256 * c;
        float in[Hh];
#pragma unroll
        for (int hh = 0; hh < Hh; ++hh) in[hh] = tile[hh * Ntok + m];
#pragma unroll
        for (int g = 0; g < Hh; ++g) {
            float acc = bls[g];
#pragma unroll
            for (int hh = 0; hh < Hh; ++hh) acc += wls[g * Hh + hh] * in[hh];
            tile[g * Ntok + m] = acc;
        }
    }
    __syncthreads();

    // softmax per row; 16 threads per row
    const int r = t >> 4, j = t & 15;
    float mx = -3.0e38f;
    for (int m = j; m < Ntok; m += 16) mx = fmaxf(mx, tile[r * Ntok + m]);
    red[r * 16 + j] = mx;
    __syncthreads();
    if (j == 0) {
        float v = red[r * 16];
        for (int q = 1; q < 16; ++q) v = fmaxf(v, red[r * 16 + q]);
        red[r * 16] = v;
    }
    __syncthreads();
    mx = red[r * 16];
    __syncthreads();
    float sum = 0.0f;
    for (int m = j; m < Ntok; m += 16) {
        float e = __expf(tile[r * Ntok + m] - mx);
        tile[r * Ntok + m] = e;
        sum += e;
    }
    red[r * 16 + j] = sum;
    __syncthreads();
    if (j == 0) {
        float v = 0.0f;
        for (int q = 0; q < 16; ++q) v += red[r * 16 + q];
        red[r * 16] = v;
    }
    __syncthreads();
    const float inv = 1.0f / red[r * 16];
    for (int m = j; m < Ntok; m += 16) tile[r * Ntok + m] *= inv;
    __syncthreads();

    // mix 2 (ww,bw): stream to global, fp32 + bf16 copies
#pragma unroll
    for (int c = 0; c < 4; ++c) {
        const int m = t + 256 * c;
        float in[Hh];
#pragma unroll
        for (int hh = 0; hh < Hh; ++hh) in[hh] = tile[hh * Ntok + m];
#pragma unroll
        for (int g = 0; g < Hh; ++g) {
            float acc = bws[g];
#pragma unroll
            for (int hh = 0; hh < Hh; ++hh) acc += wws[g * Hh + hh] * in[hh];
            basep[g * plane + m]  = acc;
            basebf[g * plane + m] = bf16_bits(acc);
        }
    }
}

// ============================================================
// V transpose: vT[b,h,d,m] = qkv_v[b,m,h,d]  (bf16), 64x64 LDS tiles
// grid (Ntok/64, 1, B*H), 256 threads (64 x 4)
// ============================================================
__global__ void __launch_bounds__(256)
v_transpose(const u16* __restrict__ qkvbf, u16* __restrict__ vT) {
    __shared__ u16 tile[64][65];
    const int bh = blockIdx.z, b = bh >> 4, h = bh & 15;
    const int m0 = blockIdx.x * 64;
    const int tx = threadIdx.x & 63, ty = threadIdx.x >> 6;  // ty: 0..3

    const u16* src = qkvbf + (size_t)b * Ntok * C3 + 2 * Cc + h * HDd;
#pragma unroll
    for (int i = 0; i < 16; ++i) {
        int mr = ty + i * 4;
        tile[mr][tx] = src[(size_t)(m0 + mr) * C3 + tx];
    }
    __syncthreads();
    u16* dst = vT + (size_t)bh * HDd * Ntok;
#pragma unroll
    for (int i = 0; i < 16; ++i) {
        int dr = ty + i * 4;
        dst[(size_t)dr * Ntok + m0 + tx] = tile[tx][dr];
    }
}

// ============================================================
// A.V: out0bf[b,n,h,d] = sum_m attnbf[b,h,n,m] * vT[b,h,d,m]
// K-dim = m = 1024. grid (16, 2, B*H). Output layout [BN, C] bf16.
// ============================================================
__global__ void __launch_bounds__(256)
av_kernel(const u16* __restrict__ attnbf, const u16* __restrict__ vT,
          u16* __restrict__ out0bf) {
    const int lane = threadIdx.x & 31;
    const int wave = threadIdx.x >> 5;
    const int bh = blockIdx.z;
    const int b = bh >> 4, h = bh & 15;

    const int n0 = blockIdx.x * 64 + (wave & 3) * 16;   // token rows (M)
    const int d0 = blockIdx.y * 32 + (wave >> 2) * 16;  // head-dim cols (N)

    const u16* ap = attnbf + (size_t)bh * Ntok * Ntok
                  + (size_t)(n0 + (lane & 15)) * Ntok + ((lane >> 4) * 16);
    const u16* bp = vT + ((size_t)bh * HDd + d0 + (lane & 15)) * Ntok
                  + ((lane >> 4) * 16);

    v8f acc = {};
#pragma unroll 4
    for (int kk = 0; kk < Ntok; kk += 32) {
        __builtin_prefetch(ap + 128, 0, 3);
        __builtin_prefetch(bp + 128, 0, 3);
        acc = WMMA_BF16(frag_ld_a(ap), frag_ld_b(bp), acc);
        ap += 32; bp += 32;
    }
    const int dc = d0 + (lane & 15);
    const int mh = (lane >> 4) * 8;
#pragma unroll
    for (int r = 0; r < 8; ++r)
        out0bf[((size_t)b * Ntok + n0 + mh + r) * Cc + h * HDd + dc] = bf16_bits(acc[r]);
}

// ============================================================
extern "C" void kernel_launch(void* const* d_in, const int* in_sizes, int n_in,
                              void* d_out, int out_size, void* d_ws, size_t ws_size,
                              hipStream_t stream) {
    (void)in_sizes; (void)n_in; (void)out_size; (void)ws_size;
    const float* x      = (const float*)d_in[0];
    const float* qkv_w  = (const float*)d_in[1];
    const float* qkv_b  = (const float*)d_in[2];
    const float* proj_w = (const float*)d_in[3];
    const float* proj_b = (const float*)d_in[4];
    const float* wl     = (const float*)d_in[5];
    const float* bl     = (const float*)d_in[6];
    const float* ww     = (const float*)d_in[7];
    const float* bw     = (const float*)d_in[8];

    float* out      = (float*)d_out;                 // [B,N,C] fp32
    float* attn_out = out + (size_t)Bb * Ntok * Cc;  // [B,H,N,N] fp32

    // ws layout (bf16 stored as u16; all chunks 16B-aligned)
    u16* xbf     = (u16*)d_ws;                             //  8.4 MB
    u16* qkvwbf  = xbf     + (size_t)BN * Cc;              //  6.3 MB
    u16* projwbf = qkvwbf  + (size_t)C3 * Cc;              //  2.1 MB
    u16* qkvbf   = projwbf + (size_t)Cc * Cc;              // 25.2 MB
    u16* vT      = qkvbf   + (size_t)BN * C3;              //  8.4 MB
    u16* out0bf  = vT      + (size_t)Bb * Hh * HDd * Ntok; //  8.4 MB
    u16* attnbf  = out0bf  + (size_t)BN * Cc;              //  134 MB

    dim3 blk(256);

    // 0) one-time fp32 -> bf16 conversions (out of all hot loops)
    f32_to_bf16_kernel<<<dim3((BN * Cc) / 1024), blk, 0, stream>>>(x, xbf);
    f32_to_bf16_kernel<<<dim3((C3 * Cc) / 1024), blk, 0, stream>>>(qkv_w, qkvwbf);
    f32_to_bf16_kernel<<<dim3((Cc * Cc) / 1024), blk, 0, stream>>>(proj_w, projwbf);

    // 1) qkv = x @ qkv_w^T + qkv_b   (bf16 out)
    gemm_bf16<Cc, true><<<dim3(BN / 64, C3 / 32), blk, 0, stream>>>(
        xbf, qkvwbf, qkv_b, qkvbf, C3);

    // 1b) transpose V for contiguous B-fragments in A.V
    v_transpose<<<dim3(Ntok / 64, 1, Bb * Hh), blk, 0, stream>>>(qkvbf, vT);

    // 2) attn0 = scale * q @ k^T -> fp32 attn region of d_out
    qk_kernel<<<dim3(Ntok / 64, Ntok / 32, Bb * Hh), blk, 0, stream>>>(
        qkvbf, attn_out);

    // 3) fused wl-mix -> softmax -> ww-mix (fp32 in place + bf16 copy)
    const size_t smem_bytes = (size_t)(Hh * Ntok + 256 + 256 + 256 + 16 + 16) * sizeof(float);
    mix_softmax_mix<<<dim3(BN), blk, smem_bytes, stream>>>(
        attn_out, attnbf, wl, bl, ww, bw);

    // 4) out0 = attn @ v  (bf16 out, [B,N,C] layout)
    av_kernel<<<dim3(Ntok / 64, HDd / 32, Bb * Hh), blk, 0, stream>>>(
        attnbf, vT, out0bf);

    // 5) out = out0 @ proj_w^T + proj_b  (fp32 -> d_out)
    gemm_bf16<Cc, false><<<dim3(BN / 64, Cc / 32), blk, 0, stream>>>(
        out0bf, projwbf, proj_b, out, Cc);
}
